// CRF_Loss_85942295593409
// MI455X (gfx1250) — matrix operand order repaired and checked by reading, not compiled, not run on hardware
//
#include <hip/hip_runtime.h>

// ---------- types ----------
typedef __attribute__((ext_vector_type(16))) _Float16 v16h;
typedef __attribute__((ext_vector_type(8)))  float    v8f;
typedef __attribute__((ext_vector_type(4)))  unsigned int v4u;

union Frag { v16h h; v4u u[2]; };

#define BB   128
#define SS   1024
#define TT   256          // TAGSET
#define NT   258          // NUM_TAGS
#define STARTT 256
#define ENDT   257

// ------------------------------------------------------------------
// Kernel 1: Tmax = max(trans[:256,:256]);  E16T[j][i] = f16(exp(T[i][j]-Tmax))
// (transposed so B-matrix WMMA fragments are contiguous 32B reads)
// ------------------------------------------------------------------
__global__ __launch_bounds__(1024)
void crf_prep(const float* __restrict__ trans,
              _Float16* __restrict__ E, float* __restrict__ tmax_out) {
    __shared__ float red[32];
    int tid = threadIdx.x;
    float mx = -3.4e38f;
    for (int f = tid; f < TT * TT; f += 1024) {
        int i = f >> 8, j = f & 255;
        mx = fmaxf(mx, trans[i * NT + j]);
    }
    #pragma unroll
    for (int o = 16; o; o >>= 1) mx = fmaxf(mx, __shfl_xor(mx, o, 32));
    if ((tid & 31) == 0) red[tid >> 5] = mx;
    __syncthreads();
    if (tid < 32) {
        float v = red[tid];
        #pragma unroll
        for (int o = 16; o; o >>= 1) v = fmaxf(v, __shfl_xor(v, o, 32));
        if (tid == 0) { red[0] = v; *tmax_out = v; }
    }
    __syncthreads();
    float Tmax = red[0];
    for (int f = tid; f < TT * TT; f += 1024) {
        int i = f >> 8, j = f & 255;
        E[j * TT + i] = (_Float16)__expf(trans[i * NT + j] - Tmax);
    }
}

// ------------------------------------------------------------------
// Kernel 2: gold-path score llh[b] and sequence length slen[b].
// One wave32 per batch row.
// ------------------------------------------------------------------
__global__ __launch_bounds__(32)
void crf_llh(const float* __restrict__ logits, const float* __restrict__ trans,
             const int* __restrict__ y,
             float* __restrict__ llh_out, float* __restrict__ slen_out) {
    int b = blockIdx.x;
    int lane = threadIdx.x;
    const int* yb = y + b * SS;
    float acc = 0.f;
    int len = 0;
    for (int t = lane; t < SS; t += 32) {
        int yt = yb[t];
        bool m = (yt != -1);
        len += m ? 1 : 0;
        if (t >= 1 && m) {
            int prev = yb[t - 1];
            int pt = prev < 0 ? 0 : prev;
            float em = __builtin_nontemporal_load(&logits[((size_t)b * SS + t) * TT + yt]);
            acc += em + trans[pt * NT + yt];
        }
    }
    #pragma unroll
    for (int o = 16; o; o >>= 1) {
        acc += __shfl_xor(acc, o, 32);
        len += __shfl_xor(len, o, 32);
    }
    if (lane == 0) {
        int y0 = yb[0];
        int t0 = y0 < 0 ? 0 : y0;
        float v = trans[STARTT * NT + t0];
        if (y0 != -1) v += logits[(size_t)b * SS * TT + t0];
        int li = len - 1; if (li < 0) li = 0;
        int yl = yb[li];
        int lt = yl < 0 ? 0 : yl;
        v += trans[lt * NT + ENDT];
        llh_out[b]  = acc + v;
        slen_out[b] = (float)len;
    }
}

// ------------------------------------------------------------------
// Kernel 3: forward scan (log-partition) via exp-domain WMMA GEMM.
// grid = 8 workgroups (16 batch rows each), block = 512 = 16 waves.
// Wave w owns output-tag stripe [16w, 16w+16); its B fragments (E16T
// stripe) stay in registers for all 1023 steps. All 8 A fragments are
// loaded in one LDS clause, then the 8 WMMAs issue back-to-back.
// ------------------------------------------------------------------
__global__ __launch_bounds__(512)
void crf_forward(const float* __restrict__ logits, const float* __restrict__ trans,
                 const int* __restrict__ y, const _Float16* __restrict__ E,
                 const float* __restrict__ tmax_p, float* __restrict__ logz_out) {
    __shared__ alignas(16) float    la[16][TT];     // log alpha, f32
    __shared__ alignas(16) _Float16 a16[16][TT];    // exp(la - m), f16
    __shared__ float mrow[16];
    __shared__ float maskrow[16];

    const int tid  = threadIdx.x;
    const int lane = tid & 31;
    const int w    = tid >> 5;      // wave id 0..15 = row (phase A) / N-tile (phase B)
    const int hh   = lane >> 4;     // lane half 0/1
    const int l16  = lane & 15;
    const int b0   = blockIdx.x * 16;
    const float Tmax = *tmax_p;

    // t = 0 init: la[b][j] = logits[b,0,j] + trans[START,j]
    for (int f = tid; f < 16 * TT; f += 512) {
        int r = f >> 8, j = f & 255;
        la[r][j] = __builtin_nontemporal_load(&logits[((size_t)(b0 + r) * SS) * TT + j])
                 + trans[STARTT * NT + j];
    }

    // Load the 8 resident B fragments for this wave's 16-column stripe.
    // B layout (16x16x32 f16): lane L -> N = L%16; lanes 0-15 hold K 0..15,
    // lanes 16-31 hold K 16..31, element e <-> K = 16*half + e (contiguous).
    Frag bf[8];
    {
        const v4u* Ebase = (const v4u*)(E + (size_t)(w * 16 + l16) * TT);
        #pragma unroll
        for (int kb = 0; kb < 8; kb++) {
            bf[kb].u[0] = Ebase[4 * kb + 2 * hh];
            bf[kb].u[1] = Ebase[4 * kb + 2 * hh + 1];
        }
    }
    __syncthreads();

    for (int t = 1; t < SS; t++) {
        // ---- Phase A: wave w normalizes row w: m = max, a16 = exp(la - m)
        float lv[8];
        float mx = -3.4e38f;
        #pragma unroll
        for (int k = 0; k < 8; k++) {
            lv[k] = la[w][k * 32 + lane];
            mx = fmaxf(mx, lv[k]);
        }
        #pragma unroll
        for (int o = 16; o; o >>= 1) mx = fmaxf(mx, __shfl_xor(mx, o, 32));
        #pragma unroll
        for (int k = 0; k < 8; k++)
            a16[w][k * 32 + lane] = (_Float16)__expf(lv[k] - mx);
        if (lane == 0) {
            mrow[w]    = mx;
            maskrow[w] = (y[(b0 + w) * SS + t] != -1) ? 1.f : 0.f;
        }
        __syncthreads();

        // ---- Phase B: wave w computes C(16x16) = A(16x256) x E_stripe(256x16)
        // A layout (16x16x32 f16): lane L -> M = L%16; elem e <-> K=16*half+e
        // (contiguous, matching B): two contiguous 16B LDS reads.
        // Load ALL fragments first (one clause, one dscnt wait), then issue
        // the 8 WMMAs back-to-back so the matrix pipe never stalls on LDS.
        Frag af[8];
        {
            const v4u* Abase = (const v4u*)&a16[l16][0];
            #pragma unroll
            for (int kb = 0; kb < 8; kb++) {
                af[kb].u[0] = Abase[4 * kb + 2 * hh];
                af[kb].u[1] = Abase[4 * kb + 2 * hh + 1];
            }
        }
        v8f c = {};
        #pragma unroll
        for (int kb = 0; kb < 8; kb++) {
            c = __builtin_amdgcn_wmma_f32_16x16x32_f16(
                    false, af[kb].h, false, bf[kb].h, (short)0, c, false, false);
        }

        // ---- Phase C: la'[b][j] = log(c) + m_b + Tmax + em, masked update.
        // D layout: element r -> M = r + 8*half, N = lane%16.
        #pragma unroll
        for (int r = 0; r < 8; r++) {
            int bl = r + 8 * hh;
            int j  = w * 16 + l16;
            float em = __builtin_nontemporal_load(
                &logits[((size_t)(b0 + bl) * SS + t) * TT + j]);
            float nv  = __logf(c[r]) + mrow[bl] + Tmax + em;
            float old = la[bl][j];
            la[bl][j] = (maskrow[bl] > 0.f) ? nv : old;
        }
        __syncthreads();
    }

    // log_z[b] = logsumexp_j(la[b][j] + trans[j,END]) ; wave w -> row w
    {
        float lv[8];
        float mx = -3.4e38f;
        #pragma unroll
        for (int k = 0; k < 8; k++) {
            int j = k * 32 + lane;
            lv[k] = la[w][j] + trans[j * NT + ENDT];
            mx = fmaxf(mx, lv[k]);
        }
        #pragma unroll
        for (int o = 16; o; o >>= 1) mx = fmaxf(mx, __shfl_xor(mx, o, 32));
        float s = 0.f;
        #pragma unroll
        for (int k = 0; k < 8; k++) s += __expf(lv[k] - mx);
        #pragma unroll
        for (int o = 16; o; o >>= 1) s += __shfl_xor(s, o, 32);
        if (lane == 0) logz_out[b0 + w] = __logf(s) + mx;
    }
}

// ------------------------------------------------------------------
// Kernel 4: loss = mean_b( -(llh[b] - logz[b]) / slen[b] )
// ------------------------------------------------------------------
__global__ __launch_bounds__(128)
void crf_final(const float* __restrict__ llh, const float* __restrict__ logz,
               const float* __restrict__ slen, float* __restrict__ out) {
    __shared__ float red[4];
    int tid = threadIdx.x;
    float loss = -(llh[tid] - logz[tid]) / slen[tid];
    #pragma unroll
    for (int o = 16; o; o >>= 1) loss += __shfl_xor(loss, o, 32);
    if ((tid & 31) == 0) red[tid >> 5] = loss;
    __syncthreads();
    if (tid == 0) out[0] = (red[0] + red[1] + red[2] + red[3]) * (1.f / BB);
}

// ------------------------------------------------------------------
extern "C" void kernel_launch(void* const* d_in, const int* in_sizes, int n_in,
                              void* d_out, int out_size, void* d_ws, size_t ws_size,
                              hipStream_t stream) {
    const float* logits = (const float*)d_in[0];   // (128,1024,256) f32
    const float* trans  = (const float*)d_in[1];   // (258,258) f32
    const int*   y      = (const int*)d_in[2];     // (128,1024) int

    char* ws = (char*)d_ws;
    _Float16* E    = (_Float16*)ws;                    // 256*256*2 = 131072 B
    float* Tmax    = (float*)(ws + 131072);            // 4 B
    float* llh     = (float*)(ws + 131072 + 256);      // 512 B
    float* slen    = (float*)(ws + 131072 + 256 + 512);
    float* logz    = (float*)(ws + 131072 + 256 + 1024);

    crf_prep   <<<1,   1024, 0, stream>>>(trans, E, Tmax);
    crf_llh    <<<BB,  32,   0, stream>>>(logits, trans, y, llh, slen);
    crf_forward<<<BB/16, 512, 0, stream>>>(logits, trans, y, E, Tmax, logz);
    crf_final  <<<1,   128,  0, stream>>>(llh, logz, slen, (float*)d_out);
}
